// SelectiveSSM_39530878992838
// MI455X (gfx1250) — compile-verified, matrix operand order checked by reading, not run on hardware
//
#include <hip/hip_runtime.h>
#include <hip/hip_bf16.h>
#include <math.h>

// ---------------- problem constants ----------------
#define DD      2048          // d_inner (= K = N of the GEMMs)
#define BB      4             // batch
#define LL      4096          // sequence length
#define NS      16            // state dim
#define MM      (BB * LL)     // GEMM M = 16384
#define DT_MIN_F 1e-4f

typedef __bf16 bf16_t;
typedef __attribute__((ext_vector_type(16))) __bf16 v16bf;
typedef __attribute__((ext_vector_type(8)))  __bf16 v8bf;
typedef __attribute__((ext_vector_type(8)))  float  v8f;
typedef __attribute__((ext_vector_type(4)))  int    v4i;

// ---- CDNA5 async global->LDS copy (ASYNCcnt-tracked), with safe fallback ----
// Probe-derived signature: (v4i addrspace(1)*, v4i addrspace(3)*, imm offset, imm cpol)
#if __has_builtin(__builtin_amdgcn_global_load_async_to_lds_b128)
#define ASYNC16(g, l)                                                        \
    __builtin_amdgcn_global_load_async_to_lds_b128(                          \
        (__attribute__((address_space(1))) v4i*)(g),                         \
        (__attribute__((address_space(3))) v4i*)(l), 0, 0)
#else
#define ASYNC16(g, l) (*(v8bf*)(l) = *(const v8bf*)(g))
#endif

#if __has_builtin(__builtin_amdgcn_s_wait_asynccnt)
#define WAIT_ASYNC(n) __builtin_amdgcn_s_wait_asynccnt(n)
#else
#define WAIT_ASYNC(n) asm volatile("s_wait_asynccnt %0" ::"n"(n) : "memory")
#endif

__device__ __forceinline__ bf16_t f2bf(float f) {
    unsigned u = __builtin_bit_cast(unsigned, f);
    unsigned r = (u + 0x7FFFu + ((u >> 16) & 1u)) >> 16;   // round-to-nearest-even
    return __builtin_bit_cast(bf16_t, (unsigned short)r);
}

__device__ __forceinline__ float softplusf(float x) {
    return log1pf(__expf(-fabsf(x))) + fmaxf(x, 0.0f);
}

// ---------------- kernel 1: f32 -> bf16 convert (u) ----------------
__global__ void cvt_f32_bf16_kernel(const float* __restrict__ in,
                                    bf16_t* __restrict__ out, int n) {
    int stride = gridDim.x * blockDim.x;
    for (int i = blockIdx.x * blockDim.x + threadIdx.x; i < n; i += stride)
        out[i] = f2bf(in[i]);
}

// ---------------- kernel 2: tiled transpose + convert (W -> W^T bf16) -----
__global__ void transpose_cvt_kernel(const float* __restrict__ in,
                                     bf16_t* __restrict__ out) {
    __shared__ float tile[32][33];
    const int tilesPerRow = DD / 32;                 // 64
    int bx = blockIdx.x % tilesPerRow;
    int by = blockIdx.x / tilesPerRow;
    int col = bx * 32 + threadIdx.x;
    for (int j = threadIdx.y; j < 32; j += blockDim.y)
        tile[j][threadIdx.x] = in[(size_t)(by * 32 + j) * DD + col];
    __syncthreads();
    int ocol = by * 32 + threadIdx.x;
    for (int j = threadIdx.y; j < 32; j += blockDim.y)
        out[(size_t)(bx * 32 + j) * DD + ocol] = f2bf(tile[threadIdx.x][j]);
}

// ---------------- kernel 3: WMMA bf16 GEMM, LDS double-buffered -----------
// Block = 256 thr (8 waves) -> 128x128 output tile. Waves arranged 4(M)x2(N);
// each wave computes 32x64 = 2x4 WMMA tiles. K staged in 32-wide chunks
// through async-copied LDS panels (padded stride 80B).
#define ROWSTRIDE 80
#define PANELBYTES (128 * ROWSTRIDE)

__global__ void gemm_wmma_bf16_lds_kernel(const bf16_t* __restrict__ Abuf,
                                          const bf16_t* __restrict__ Wt,
                                          const float*  __restrict__ bias,
                                          float* __restrict__ out, int mode) {
    const int K = DD, NC = DD;
    __shared__ __align__(16) char lds[2][2][PANELBYTES];   // [buf][A/B][panel]

    const int lane   = threadIdx.x & 31;
    const int waveId = threadIdx.x >> 5;
    const int waveM  = waveId >> 1;                  // 0..3
    const int waveN  = waveId & 1;                   // 0..1
    const int blockRow = blockIdx.x / (NC / 128);
    const int blockCol = blockIdx.x % (NC / 128);

    const bf16_t* aPanel = Abuf + (size_t)blockRow * 128 * K;
    const bf16_t* bPanel = Wt   + (size_t)blockCol * 128 * K;

    v8f acc[2][4];
#pragma unroll
    for (int mt = 0; mt < 2; ++mt)
#pragma unroll
        for (int s = 0; s < 4; ++s)
            acc[mt][s] = (v8f){0.f,0.f,0.f,0.f,0.f,0.f,0.f,0.f};

    // issue one 32-wide K chunk: 2 panels x 8KB, 4 x 16B per thread
    auto issue = [&](int k0, int buf) {
        char* lA = &lds[buf][0][0];
        char* lB = &lds[buf][1][0];
#pragma unroll
        for (int i = 0; i < 2; ++i) {
            int lin = threadIdx.x + i * 256;         // 0..511
            int row = lin >> 2, seg = lin & 3;
            ASYNC16(aPanel + (size_t)row * K + (k0 + seg * 8),
                    lA + row * ROWSTRIDE + seg * 16);
        }
#pragma unroll
        for (int i = 0; i < 2; ++i) {
            int lin = threadIdx.x + i * 256;
            int row = lin >> 2, seg = lin & 3;
            ASYNC16(bPanel + (size_t)row * K + (k0 + seg * 8),
                    lB + row * ROWSTRIDE + seg * 16);
        }
    };

    issue(0, 0);
    const int kb = (lane >> 4) * 16;                 // lane K-striping (bytes)

    for (int kk = 0; kk < K / 32; ++kk) {
        int cur = kk & 1;
        __syncthreads();                             // prev compute on cur^1 done
        bool more = (kk + 1 < K / 32);
        if (more) { issue((kk + 1) * 32, cur ^ 1); WAIT_ASYNC(4); }
        else      { WAIT_ASYNC(0); }
        __syncthreads();                             // everyone's cur panel ready

        const char* lA = &lds[cur][0][0];
        const char* lB = &lds[cur][1][0];

        v16bf aF[2];
#pragma unroll
        for (int mt = 0; mt < 2; ++mt) {
            int r = waveM * 32 + mt * 16 + (lane & 15);
            v8bf lo = *(const v8bf*)(lA + r * ROWSTRIDE + kb);
            v8bf hi = *(const v8bf*)(lA + r * ROWSTRIDE + kb + 32);
            aF[mt] = __builtin_shufflevector(lo, hi,
                        0,1,2,3,4,5,6,7,8,9,10,11,12,13,14,15);
        }
#pragma unroll
        for (int s = 0; s < 4; ++s) {
            int r = waveN * 64 + s * 16 + (lane & 15);
            v8bf lo = *(const v8bf*)(lB + r * ROWSTRIDE + kb);
            v8bf hi = *(const v8bf*)(lB + r * ROWSTRIDE + kb + 32);
            v16bf bF = __builtin_shufflevector(lo, hi,
                        0,1,2,3,4,5,6,7,8,9,10,11,12,13,14,15);
#pragma unroll
            for (int mt = 0; mt < 2; ++mt)
                acc[mt][s] = __builtin_amdgcn_wmma_f32_16x16x32_bf16(
                    false, aF[mt], false, bF, (short)0, acc[mt][s], false, false);
        }
    }

    // Epilogue: C layout -> lane<16: M=r, N=lane; lane>=16: M=8+r, N=lane-16
    const int rowHalf = (lane >> 4) ? 8 : 0;
#pragma unroll
    for (int mt = 0; mt < 2; ++mt) {
#pragma unroll
        for (int s = 0; s < 4; ++s) {
            int col = blockCol * 128 + waveN * 64 + s * 16 + (lane & 15);
            float bv = bias[col];
            size_t rbase = (size_t)(blockRow * 128 + waveM * 32 + mt * 16 + rowHalf);
#pragma unroll
            for (int r = 0; r < 8; ++r) {
                float v = acc[mt][s][r] + bv;
                if (mode == 1) v = softplusf(v) + DT_MIN_F;
                out[(rbase + r) * NC + col] = v;
            }
        }
    }
}

// ---------------- kernel 4: selective-scan over L -------------------------
__global__ void ssm_scan_kernel(const float* __restrict__ u,
                                const float* __restrict__ dt,
                                const float* __restrict__ bt,
                                const float* __restrict__ ct,
                                const float* __restrict__ a_raw,
                                const float* __restrict__ B_base,
                                const float* __restrict__ C_base,
                                const float* __restrict__ Dvec,
                                float* __restrict__ carry,
                                float* __restrict__ y) {
    int gid  = blockIdx.x * blockDim.x + threadIdx.x;
    int chan = gid >> 4;
    int n    = gid & 15;
    if (chan >= BB * DD) return;
    int b  = chan / DD;
    int dc = chan % DD;

    float An = -softplusf(a_raw[dc * NS + n]);
    float Bn = B_base[dc * NS + n];
    float Cn = C_base[dc * NS + n];
    float Dv = Dvec[dc];

    float x = 0.0f;
    size_t base = ((size_t)b * LL) * DD + dc;
    for (int t = 0; t < LL; ++t) {
        size_t idx = base + (size_t)t * DD;
        float us  = u[idx];
        float dts = dt[idx];
        float bs  = bt[idx];
        float cs  = ct[idx];
        float Abar = __expf(dts * An);
        x = Abar * x + (dts * bs * Bn) * us;
        float r = cs * Cn * x;
        r += __shfl_xor(r, 1, 16);
        r += __shfl_xor(r, 2, 16);
        r += __shfl_xor(r, 4, 16);
        r += __shfl_xor(r, 8, 16);
        if (n == 0) y[idx] = r + Dv * us;
    }
    carry[(size_t)chan * NS + n] = x;
}

// ---------------- host launch --------------------------------------------
extern "C" void kernel_launch(void* const* d_in, const int* in_sizes, int n_in,
                              void* d_out, int out_size, void* d_ws, size_t ws_size,
                              hipStream_t stream) {
    const float* u     = (const float*)d_in[0];
    const float* Wdt   = (const float*)d_in[1];
    const float* bdt   = (const float*)d_in[2];
    const float* Wb    = (const float*)d_in[3];
    const float* bb    = (const float*)d_in[4];
    const float* Wc    = (const float*)d_in[5];
    const float* bc    = (const float*)d_in[6];
    const float* a_raw = (const float*)d_in[7];
    const float* Bbase = (const float*)d_in[8];
    const float* Cbase = (const float*)d_in[9];
    const float* Dvec  = (const float*)d_in[10];

    char* ws = (char*)d_ws;
    const size_t uElems = (size_t)MM * DD;
    const size_t wElems = (size_t)DD * DD;
    bf16_t* u_bf  = (bf16_t*)(ws);
    bf16_t* WtDt  = (bf16_t*)(ws + uElems * 2);
    bf16_t* WtB   = (bf16_t*)(ws + uElems * 2 + wElems * 2);
    bf16_t* WtC   = (bf16_t*)(ws + uElems * 2 + wElems * 4);
    float*  dtBuf = (float*)(ws + uElems * 2 + wElems * 6);
    float*  btBuf = dtBuf + uElems;
    float*  ctBuf = btBuf + uElems;

    float* carryOut = (float*)d_out;
    float* yOut     = carryOut + (size_t)BB * DD * NS;

    // 1) u -> bf16
    cvt_f32_bf16_kernel<<<32768, 256, 0, stream>>>(u, u_bf, (int)uElems);

    // 2) W -> W^T bf16 (x3)
    dim3 tb(32, 8);
    transpose_cvt_kernel<<<(DD/32)*(DD/32), tb, 0, stream>>>(Wdt, WtDt);
    transpose_cvt_kernel<<<(DD/32)*(DD/32), tb, 0, stream>>>(Wb,  WtB);
    transpose_cvt_kernel<<<(DD/32)*(DD/32), tb, 0, stream>>>(Wc,  WtC);

    // 3) three WMMA GEMMs: (M/128)*(NC/128) = 128*16 = 2048 blocks
    const int gemmBlocks = (MM / 128) * (DD / 128);
    gemm_wmma_bf16_lds_kernel<<<gemmBlocks, 256, 0, stream>>>(u_bf, WtDt, bdt, dtBuf, 1);
    gemm_wmma_bf16_lds_kernel<<<gemmBlocks, 256, 0, stream>>>(u_bf, WtB,  bb,  btBuf, 0);
    gemm_wmma_bf16_lds_kernel<<<gemmBlocks, 256, 0, stream>>>(u_bf, WtC,  bc,  ctBuf, 0);

    // 4) scan
    ssm_scan_kernel<<<(BB * DD * NS) / 256, 256, 0, stream>>>(
        u, dtBuf, btBuf, ctBuf, a_raw, Bbase, Cbase, Dvec, carryOut, yOut);
}
